// Block_29343216566583
// MI455X (gfx1250) — matrix operand order repaired and checked by reading, not compiled, not run on hardware
//
#include <hip/hip_runtime.h>
#include <hip/hip_bf16.h>
#include <math.h>

typedef __attribute__((ext_vector_type(16))) _Float16 v16h;
typedef __attribute__((ext_vector_type(8)))  _Float16 v8h;
typedef __attribute__((ext_vector_type(8)))  float    v8f;

#define SEQ   1024
#define DIM   768
#define NH    12
#define HD    64
#define DMLP  3072
#define BATCH 16
#define BN    (BATCH*SEQ)
#define ATT_SCALE 0.125f   // 1/sqrt(64)

union AFrag { v16h v; v8h h[2]; };

static __device__ __forceinline__ v8f wmma_f16(v16h a, v16h b, v8f c) {
  // D = A(16x32 f16) * B(32x16 f16) + C(16x16 f32)
  return __builtin_amdgcn_wmma_f32_16x16x32_f16(false, a, false, b, (short)0, c,
                                                false, false);
}

// Async 16-byte global->LDS copy (CDNA5, tracked by ASYNCcnt).
// ldsaddr: LDS byte address (low 32 bits of flat shared pointer).
static __device__ __forceinline__ void async_cp16(unsigned ldsaddr, const void* gaddr) {
  asm volatile("global_load_async_to_lds_b128 %0, %1, off"
               :: "v"(ldsaddr), "v"(gaddr) : "memory");
}
static __device__ __forceinline__ void wait_async0() {
  asm volatile("s_wait_asynccnt 0x0" ::: "memory");
}

// ---------------------------------------------------------------------------
// Weight pack: fp32 W[K x N] row-major -> f16 B-fragment stream.
// Packed layout: [kt][nt][lane 0..31][elem 0..15] with
//   lane<16 : element i -> W[kt*32 + i     ][nt*16 + lane]
//   lane>=16: element i -> W[kt*32 + 16 + i][nt*16 + lane-16]
// ---------------------------------------------------------------------------
__global__ __launch_bounds__(256)
void pack_w_kernel(const float* __restrict__ W, _Float16* __restrict__ P,
                   int K, int N) {
  size_t idx = (size_t)blockIdx.x * 256 + threadIdx.x;
  int i = (int)(idx & 15);
  int l = (int)((idx >> 4) & 31);
  size_t tile = idx >> 9;
  int ntiles = N >> 4;
  int nt = (int)(tile % ntiles);
  int kt = (int)(tile / ntiles);
  int k = kt * 32 + ((l < 16) ? 0 : 16) + i;
  int n = nt * 16 + (l & 15);
  P[idx] = (_Float16)W[(size_t)k * N + n];
}

// ---------------------------------------------------------------------------
// LayerNorm: fp32 in -> f16 out (one row of D=768 per 256-thread block)
// ---------------------------------------------------------------------------
__global__ __launch_bounds__(256)
void ln_kernel(const float* __restrict__ x, const float* __restrict__ g,
               const float* __restrict__ bt, _Float16* __restrict__ out) {
  __shared__ float s1[256];
  __shared__ float s2[256];
  int row = blockIdx.x;
  const float* xr = x + (size_t)row * DIM;
  float vals[3];
  float lsum = 0.f, lsq = 0.f;
#pragma unroll
  for (int i = 0; i < 3; ++i) {
    float v = xr[threadIdx.x + i * 256];
    vals[i] = v;
    lsum += v; lsq += v * v;
  }
  s1[threadIdx.x] = lsum; s2[threadIdx.x] = lsq;
  __syncthreads();
  for (int st = 128; st > 0; st >>= 1) {
    if (threadIdx.x < st) {
      s1[threadIdx.x] += s1[threadIdx.x + st];
      s2[threadIdx.x] += s2[threadIdx.x + st];
    }
    __syncthreads();
  }
  float mean = s1[0] * (1.0f / DIM);
  float var  = s2[0] * (1.0f / DIM) - mean * mean;
  float rstd = rsqrtf(var + 1e-6f);
#pragma unroll
  for (int i = 0; i < 3; ++i) {
    int c = threadIdx.x + i * 256;
    out[(size_t)row * DIM + c] = (_Float16)((vals[i] - mean) * rstd * g[c] + bt[c]);
  }
}

// ---------------------------------------------------------------------------
// WMMA GEMM: C[M x N] = A(f16 row-major, lda=K) * Bpacked(f16) + bias
// Block = 4 waves (128 thr). Block tile 128(M) x 64(N); wave w owns rows
// [by*128 + w*32, +32), all 64 cols -> 8 accumulators, 8 WMMA / k-step.
// B k-slice (4 KB) staged in LDS via double-buffered async copies (ASYNCcnt),
// shared by all 4 waves.
// ---------------------------------------------------------------------------
enum { EPI_QK = 0, EPI_V = 1, EPI_RESF32 = 2, EPI_GELU = 3 };

template <int EPI>
__global__ __launch_bounds__(128)
void gemm_kernel(const _Float16* __restrict__ A, const _Float16* __restrict__ Bp,
                 const float* __restrict__ bias, const float* __restrict__ res,
                 void* __restrict__ out, int M, int N, int K) {
  __shared__ _Float16 sb[2][2048];   // 2 x 4KB B staging buffers
  const int tid  = threadIdx.x;
  const int lane = tid & 31;
  const int wave = tid >> 5;
  const int row0 = blockIdx.y * 128 + wave * 32;
  const int col0 = blockIdx.x * 64;
  const int ntiles = N >> 4;
  const size_t bkstep = (size_t)ntiles * 512;            // halves per kt step
  const _Float16* bsrc = Bp + (size_t)(col0 >> 4) * 512; // + kt*bkstep
  const int nk = K >> 5;

  v8f acc[8];
#pragma unroll
  for (int t = 0; t < 8; ++t) {
    v8f z = {0.f, 0.f, 0.f, 0.f, 0.f, 0.f, 0.f, 0.f};
    acc[t] = z;
  }

  const int arow = lane & 15;
  const int akb  = (lane < 16) ? 0 : 16;
  const _Float16* aptr0 = A + (size_t)(row0 + arow) * K + akb;
  const _Float16* aptr1 = A + (size_t)(row0 + 16 + arow) * K + akb;

  // prologue: stage kt=0 into buffer 0 (4KB: 128 threads x 2 x 16B)
  {
    unsigned lb = (unsigned)(uintptr_t)(&sb[0][0]);
    const char* src = (const char*)bsrc;
    async_cp16(lb + tid * 16,        src + tid * 16);
    async_cp16(lb + tid * 16 + 2048, src + tid * 16 + 2048);
  }

  for (int kt = 0; kt < nk; ++kt) {
    wait_async0();       // own async copies of slice kt complete
    __syncthreads();     // all waves' copies complete; prior reads of the
                         // buffer we are about to overwrite have retired
    if (kt + 1 < nk) {
      unsigned lb = (unsigned)(uintptr_t)(&sb[(kt + 1) & 1][0]);
      const char* src = (const char*)(bsrc + (size_t)(kt + 1) * bkstep);
      async_cp16(lb + tid * 16,        src + tid * 16);
      async_cp16(lb + tid * 16 + 2048, src + tid * 16 + 2048);
    }
    const int k0 = kt * 32;
    AFrag fa0, fa1;
    fa0.h[0] = *(const v8h*)(aptr0 + k0);
    fa0.h[1] = *(const v8h*)(aptr0 + k0 + 8);
    fa1.h[0] = *(const v8h*)(aptr1 + k0);
    fa1.h[1] = *(const v8h*)(aptr1 + k0 + 8);
    const _Float16* bb = &sb[kt & 1][lane * 16];
    v16h b0 = *(const v16h*)(bb);
    v16h b1 = *(const v16h*)(bb + 512);
    v16h b2 = *(const v16h*)(bb + 1024);
    v16h b3 = *(const v16h*)(bb + 1536);
    acc[0] = wmma_f16(fa0.v, b0, acc[0]);
    acc[1] = wmma_f16(fa0.v, b1, acc[1]);
    acc[2] = wmma_f16(fa0.v, b2, acc[2]);
    acc[3] = wmma_f16(fa0.v, b3, acc[3]);
    acc[4] = wmma_f16(fa1.v, b0, acc[4]);
    acc[5] = wmma_f16(fa1.v, b1, acc[5]);
    acc[6] = wmma_f16(fa1.v, b2, acc[6]);
    acc[7] = wmma_f16(fa1.v, b3, acc[7]);
  }

  // Epilogue. C layout: lane<16 vgpr v -> C[M=v][N=lane]; lane>=16 -> C[M=8+v][N=lane-16]
  const int cl   = lane & 15;
  const int rofs = (lane < 16) ? 0 : 8;
#pragma unroll
  for (int mt = 0; mt < 2; ++mt) {
#pragma unroll
    for (int t = 0; t < 4; ++t) {
      int col = col0 + t * 16 + cl;
      float bv = bias[col];
#pragma unroll
      for (int v = 0; v < 8; ++v) {
        int r = row0 + mt * 16 + rofs + v;
        float val = acc[mt * 4 + t][v] + bv;
        if constexpr (EPI == EPI_QK) {
          int bb2 = r >> 10, tp = r & 1023;
          int hh = col >> 6, hd = col & 63;
          ((_Float16*)out)[(((size_t)(bb2 * NH + hh)) * SEQ + tp) * HD + hd] = (_Float16)val;
        } else if constexpr (EPI == EPI_V) {
          int bb2 = r >> 10, tp = r & 1023;
          int hh = col >> 6, hd = col & 63;
          ((_Float16*)out)[(((size_t)(bb2 * NH + hh)) * HD + hd) * SEQ + tp] = (_Float16)val;
        } else if constexpr (EPI == EPI_RESF32) {
          ((float*)out)[(size_t)r * N + col] = val + res[(size_t)r * N + col];
        } else {  // EPI_GELU (exact erf gelu)
          float gl = 0.5f * val * (1.0f + erff(val * 0.70710678118654752f));
          ((_Float16*)out)[(size_t)r * N + col] = (_Float16)gl;
        }
      }
    }
  }
}

// ---------------------------------------------------------------------------
// Flash attention: one wave (32 thr) per (b, h, 16-query tile).
// q: [B,H,N,64] f16 (A-operand), k: [B,H,N,64] f16 (row-major == K^T B-operand),
// vt: [B,H,64,N] f16 (transposed == V B-operand). ctx out: f16 [BN, DIM] row-major.
// ---------------------------------------------------------------------------
__global__ __launch_bounds__(32)
void attn_kernel(const _Float16* __restrict__ qf, const _Float16* __restrict__ kf,
                 const _Float16* __restrict__ vt, _Float16* __restrict__ ctx) {
  __shared__ _Float16 lds[16 * 32];   // probs tile, C-layout -> A-layout bounce
  const int lane = threadIdx.x;
  const int bid = blockIdx.x;
  const int qt = bid & 63;                 // SEQ/16 tiles
  const int h  = (bid >> 6) % NH;
  const int b  = bid / (64 * NH);

  const _Float16* qbase = qf + (((size_t)(b * NH + h)) * SEQ + qt * 16) * HD;
  const _Float16* kbase = kf + ((size_t)(b * NH + h)) * SEQ * HD;
  const _Float16* vbase = vt + ((size_t)(b * NH + h)) * HD * SEQ;

  const int arow = lane & 15;
  const int akb  = (lane < 16) ? 0 : 16;
  AFrag aq0, aq1;
  aq0.h[0] = *(const v8h*)(qbase + arow * HD + akb);
  aq0.h[1] = *(const v8h*)(qbase + arow * HD + akb + 8);
  aq1.h[0] = *(const v8h*)(qbase + arow * HD + 32 + akb);
  aq1.h[1] = *(const v8h*)(qbase + arow * HD + 32 + akb + 8);

  float m[8], s[8];
  v8f acc[4];
#pragma unroll
  for (int v = 0; v < 8; ++v) { m[v] = -1e30f; s[v] = 0.f; }
#pragma unroll
  for (int t = 0; t < 4; ++t) {
    v8f z = {0.f, 0.f, 0.f, 0.f, 0.f, 0.f, 0.f, 0.f};
    acc[t] = z;
  }

  const int bn    = lane & 15;
  const int bkoff = (lane < 16) ? 0 : 16;

  for (int j = 0; j < SEQ; j += 32) {
    // ---- scores: S[16q x 32k] = Q * K^T, 2 key tiles x 2 hd chunks ----
    const _Float16* kb0 = kbase + (size_t)j * HD + bn * HD;
    const _Float16* kb1 = kb0 + 16 * HD;
    AFrag bk00, bk01, bk10, bk11;
    bk00.h[0] = *(const v8h*)(kb0 + bkoff);       bk00.h[1] = *(const v8h*)(kb0 + bkoff + 8);
    bk01.h[0] = *(const v8h*)(kb0 + 32 + bkoff);  bk01.h[1] = *(const v8h*)(kb0 + 32 + bkoff + 8);
    bk10.h[0] = *(const v8h*)(kb1 + bkoff);       bk10.h[1] = *(const v8h*)(kb1 + bkoff + 8);
    bk11.h[0] = *(const v8h*)(kb1 + 32 + bkoff);  bk11.h[1] = *(const v8h*)(kb1 + 32 + bkoff + 8);
    v8f z0 = {0.f, 0.f, 0.f, 0.f, 0.f, 0.f, 0.f, 0.f};
    v8f s0 = z0, s1 = z0;
    s0 = wmma_f16(aq0.v, bk00.v, s0);
    s0 = wmma_f16(aq1.v, bk01.v, s0);
    s1 = wmma_f16(aq0.v, bk10.v, s1);
    s1 = wmma_f16(aq1.v, bk11.v, s1);

    // ---- online softmax (rows in VGPRs, cols across 16-lane halves) ----
    float p0[8], p1[8], alpha[8];
#pragma unroll
    for (int v = 0; v < 8; ++v) {
      float x0 = s0[v] * ATT_SCALE;
      float x1 = s1[v] * ATT_SCALE;
      float cm = fmaxf(x0, x1);
      cm = fmaxf(cm, __shfl_xor(cm, 1));
      cm = fmaxf(cm, __shfl_xor(cm, 2));
      cm = fmaxf(cm, __shfl_xor(cm, 4));
      cm = fmaxf(cm, __shfl_xor(cm, 8));
      float mn = fmaxf(m[v], cm);
      float al = __expf(m[v] - mn);
      m[v] = mn;
      float e0 = __expf(x0 - mn);
      float e1 = __expf(x1 - mn);
      float rs = e0 + e1;
      rs += __shfl_xor(rs, 1);
      rs += __shfl_xor(rs, 2);
      rs += __shfl_xor(rs, 4);
      rs += __shfl_xor(rs, 8);
      s[v] = s[v] * al + rs;
      alpha[v] = al;
      p0[v] = e0; p1[v] = e1;
    }
#pragma unroll
    for (int t = 0; t < 4; ++t)
#pragma unroll
      for (int v = 0; v < 8; ++v) acc[t][v] *= alpha[v];

    // ---- probs C-layout -> LDS -> A-layout fragment ----
    __syncthreads();
    {
      int prow = (lane < 16) ? 0 : 8;
#pragma unroll
      for (int v = 0; v < 8; ++v) {
        lds[(prow + v) * 32 + bn]      = (_Float16)p0[v];
        lds[(prow + v) * 32 + bn + 16] = (_Float16)p1[v];
      }
    }
    __syncthreads();
    AFrag pa;
    pa.h[0] = *(const v8h*)(&lds[arow * 32 + akb]);
    pa.h[1] = *(const v8h*)(&lds[arow * 32 + akb + 8]);

    // ---- ctx += P(16x32) * V(32x64) ----
#pragma unroll
    for (int t = 0; t < 4; ++t) {
      const _Float16* vb = vbase + (size_t)(t * 16 + bn) * SEQ + j + bkoff;
      AFrag fb;
      fb.h[0] = *(const v8h*)(vb);
      fb.h[1] = *(const v8h*)(vb + 8);
      acc[t] = wmma_f16(pa.v, fb.v, acc[t]);
    }
    __syncthreads();
  }

  // ---- normalize and scatter to [BN, DIM] row-major f16 ----
  const int rofs = (lane < 16) ? 0 : 8;
  const int cl   = lane & 15;
#pragma unroll
  for (int t = 0; t < 4; ++t) {
#pragma unroll
    for (int v = 0; v < 8; ++v) {
      int tp = qt * 16 + rofs + v;
      int col = h * HD + t * 16 + cl;
      ctx[((size_t)(b * SEQ + tp)) * DIM + col] = (_Float16)(acc[t][v] / s[v]);
    }
  }
}

// ---------------------------------------------------------------------------
// Launch
// ---------------------------------------------------------------------------
extern "C" void kernel_launch(void* const* d_in, const int* in_sizes, int n_in,
                              void* d_out, int out_size, void* d_ws, size_t ws_size,
                              hipStream_t stream) {
  const float* x     = (const float*)d_in[0];
  const float* wq    = (const float*)d_in[1];
  const float* bq    = (const float*)d_in[2];
  const float* wk    = (const float*)d_in[3];
  const float* bk    = (const float*)d_in[4];
  const float* wv    = (const float*)d_in[5];
  const float* bv    = (const float*)d_in[6];
  const float* wo    = (const float*)d_in[7];
  const float* bo    = (const float*)d_in[8];
  const float* w1    = (const float*)d_in[9];
  const float* b1    = (const float*)d_in[10];
  const float* w2    = (const float*)d_in[11];
  const float* b2    = (const float*)d_in[12];
  const float* ln1g  = (const float*)d_in[13];
  const float* ln1b  = (const float*)d_in[14];
  const float* ln2g  = (const float*)d_in[15];
  const float* ln2b  = (const float*)d_in[16];
  float* out = (float*)d_out;

  // workspace carve-up (bytes)
  char* ws = (char*)d_ws;
  const size_t SZ_WDD  = (size_t)DIM * DIM * 2;
  const size_t SZ_WDM  = (size_t)DIM * DMLP * 2;
  const size_t SZ_ACT  = (size_t)BN * DIM * 2;
  const size_t SZ_ACT4 = (size_t)BN * DIM * 4;
  const size_t SZ_MLP  = (size_t)BN * DMLP * 2;

  _Float16* wq_p = (_Float16*)(ws);
  _Float16* wk_p = (_Float16*)(ws + SZ_WDD);
  _Float16* wv_p = (_Float16*)(ws + 2 * SZ_WDD);
  _Float16* wo_p = (_Float16*)(ws + 3 * SZ_WDD);
  _Float16* w1_p = (_Float16*)(ws + 4 * SZ_WDD);
  _Float16* w2_p = (_Float16*)(ws + 4 * SZ_WDD + SZ_WDM);
  char* p = ws + 4 * SZ_WDD + 2 * SZ_WDM;
  _Float16* hf   = (_Float16*)(p);              p += SZ_ACT;   // LN1 out / LN2 out (reused)
  _Float16* qf   = (_Float16*)(p);              p += SZ_ACT;
  _Float16* kf   = (_Float16*)(p);              p += SZ_ACT;
  _Float16* vtf  = (_Float16*)(p);              p += SZ_ACT;
  _Float16* ctxf = (_Float16*)(p);              p += SZ_ACT;
  float*    x1   = (float*)(p);                 p += SZ_ACT4;
  _Float16* h1f  = (_Float16*)(p);              p += SZ_MLP;

  // 1) pack weights fp32 -> f16 B-fragment layout
  pack_w_kernel<<<(DIM * DIM) / 256, 256, 0, stream>>>(wq, wq_p, DIM, DIM);
  pack_w_kernel<<<(DIM * DIM) / 256, 256, 0, stream>>>(wk, wk_p, DIM, DIM);
  pack_w_kernel<<<(DIM * DIM) / 256, 256, 0, stream>>>(wv, wv_p, DIM, DIM);
  pack_w_kernel<<<(DIM * DIM) / 256, 256, 0, stream>>>(wo, wo_p, DIM, DIM);
  pack_w_kernel<<<(DIM * DMLP) / 256, 256, 0, stream>>>(w1, w1_p, DIM, DMLP);
  pack_w_kernel<<<(DMLP * DIM) / 256, 256, 0, stream>>>(w2, w2_p, DMLP, DIM);

  // 2) LN1: x -> hf (f16)
  ln_kernel<<<BN, 256, 0, stream>>>(x, ln1g, ln1b, hf);

  // 3) QKV projections (WMMA + async-LDS staged B)
  dim3 gDD(DIM / 64, BN / 128);
  gemm_kernel<EPI_QK><<<gDD, 128, 0, stream>>>(hf, wq_p, bq, nullptr, qf, BN, DIM, DIM);
  gemm_kernel<EPI_QK><<<gDD, 128, 0, stream>>>(hf, wk_p, bk, nullptr, kf, BN, DIM, DIM);
  gemm_kernel<EPI_V ><<<gDD, 128, 0, stream>>>(hf, wv_p, bv, nullptr, vtf, BN, DIM, DIM);

  // 4) flash attention (WMMA)
  attn_kernel<<<BATCH * NH * (SEQ / 16), 32, 0, stream>>>(qf, kf, vtf, ctxf);

  // 5) output projection + residual -> x1 (f32)
  gemm_kernel<EPI_RESF32><<<gDD, 128, 0, stream>>>(ctxf, wo_p, bo, x, x1, BN, DIM, DIM);

  // 6) LN2: x1 -> hf (f16, reuse)
  ln_kernel<<<BN, 256, 0, stream>>>(x1, ln2g, ln2b, hf);

  // 7) MLP1 + exact GELU -> h1f (f16)
  dim3 gDM(DMLP / 64, BN / 128);
  gemm_kernel<EPI_GELU><<<gDM, 128, 0, stream>>>(hf, w1_p, b1, nullptr, h1f, BN, DMLP, DIM);

  // 8) MLP2 + bias + residual -> out (f32)
  gemm_kernel<EPI_RESF32><<<gDD, 128, 0, stream>>>(h1f, w2_p, b2, x1, out, BN, DIM, DMLP);
}
